// GeneralRNN_19610820674236
// MI455X (gfx1250) — matrix operand order repaired and checked by reading, not compiled, not run on hardware
//
#include <hip/hip_runtime.h>
#include <hip/hip_bf16.h>
#include <stdint.h>

// Problem sizes (from the reference): T=2048, B=64, I=H=256, 3H=768.
#define T_STEPS 2048
#define BATCH   64
#define IDIM    256
#define HDIM    256
#define GDIM    768
#define MROWS   (T_STEPS * BATCH)

// Workspace layout (bytes). Total ~258 MB.
#define WS_WIH0B   0ull
#define WS_WHH0B   393216ull
#define WS_WIH1B   786432ull
#define WS_WHH1B   1179648ull
#define WS_GI1     1572864ull          // 64*768 bf16   =    98304 B
#define WS_HL0B    1671168ull          // 64*256 bf16   =    32768 B
#define WS_HL0F    1703936ull          // 64*256 f32    =    65536 B
#define WS_GI0     1769472ull          // 2048*64*768 bf16 = 201326592 B
#define WS_XB      203096064ull        // 2048*64*256 bf16 = 67108864 B

// LDS layout inside the persistent recurrence kernel (exactly 320 KB):
//   [0,196608)        gh pre-activations, 64x768 f32
//   [196608,229376)   h state, 64x256 bf16
//   [229376,327680)   gi staging row, 64x768 bf16 (TDM destination)
#define LDS_GH_OFF 0u
#define LDS_H_OFF  196608u
#define LDS_GI_OFF 229376u
#define LDS_BYTES  327680u

typedef __attribute__((ext_vector_type(16))) __bf16 v16bf;
typedef __attribute__((ext_vector_type(8)))  float  v8f;

#if defined(__has_builtin)
# if __has_builtin(__builtin_amdgcn_tensor_load_to_lds) && __has_builtin(__builtin_amdgcn_s_wait_tensorcnt)
#  define HAVE_TDM 1
# endif
#endif
#ifndef HAVE_TDM
# define HAVE_TDM 0
#endif

typedef __attribute__((ext_vector_type(4))) unsigned tdm_v4u;
typedef __attribute__((ext_vector_type(8))) int      tdm_v8i;
typedef __attribute__((ext_vector_type(4))) int      tdm_v4i;

__device__ __forceinline__ unsigned short f2bf(float f) {
  union { float f; unsigned u; } v; v.f = f;
  unsigned u = v.u;
  u += 0x7fffu + ((u >> 16) & 1u);            // round-to-nearest-even
  return (unsigned short)(u >> 16);
}
__device__ __forceinline__ unsigned pack2bf(float a, float b) {
  return (unsigned)f2bf(a) | ((unsigned)f2bf(b) << 16);
}
__device__ __forceinline__ float bf2f(unsigned short s) {
  union { unsigned u; float f; } v; v.u = ((unsigned)s) << 16;
  return v.f;
}
__device__ __forceinline__ float sigmoid_fast(float x) {
  return 1.0f / (1.0f + __expf(-x));
}
__device__ __forceinline__ float tanh_fast(float x) {
  return 2.0f / (1.0f + __expf(-2.0f * x)) - 1.0f;   // == tanh(x)
}

union FragAB { v16bf v; unsigned short s[16]; uint4 q[2]; };

__device__ __forceinline__ v8f wmma_bf16(const FragAB& a, const FragAB& b, v8f c) {
  return __builtin_amdgcn_wmma_f32_16x16x32_bf16(false, a.v, false, b.v,
                                                 (short)0, c, false, false);
}

// ---------------------------------------------------------------------------
// Kernel 1a: convert the four 768x256 f32 weight matrices to bf16 (row-major).
// ---------------------------------------------------------------------------
__global__ void wcvt_kernel(const float* __restrict__ w0, const float* __restrict__ w1,
                            const float* __restrict__ w2, const float* __restrict__ w3,
                            unsigned short* __restrict__ o0, unsigned short* __restrict__ o1,
                            unsigned short* __restrict__ o2, unsigned short* __restrict__ o3,
                            int n) {
  int i = blockIdx.x * blockDim.x + threadIdx.x;
  if (i < n) {
    o0[i] = f2bf(w0[i]);
    o1[i] = f2bf(w1[i]);
    o2[i] = f2bf(w2[i]);
    o3[i] = f2bf(w3[i]);
  }
}

// ---------------------------------------------------------------------------
// Kernel 1b: stream-convert x (T*B*I f32) to bf16, 4 elements per thread.
// Pure bandwidth: ~200 MB total traffic, ~9 us at HBM peak.
// ---------------------------------------------------------------------------
__global__ void xcvt_kernel(const float* __restrict__ x,
                            unsigned short* __restrict__ xb, int n4) {
  int i = blockIdx.x * blockDim.x + threadIdx.x;
  if (i < n4) {
    float4 f = ((const float4*)x)[i];
    uint2 p;
    p.x = pack2bf(f.x, f.y);
    p.y = pack2bf(f.z, f.w);
    ((uint2*)xb)[i] = p;
  }
}

// ---------------------------------------------------------------------------
// Kernel 2: Gi = A @ W^T + bias. A is [M][256] bf16, W [768][256] bf16,
// output [M][768] bf16. Block = 256 (8 waves); each wave computes a 16x96
// strip (6 N-tiles, A fragment reused 6x per K-step). Grid = (M/16, 1).
// Pure global_load_b128 + v_wmma: no conversion ALU in the inner loop.
// ---------------------------------------------------------------------------
__global__ __launch_bounds__(256)
void gi_gemm_kernel(const unsigned short* __restrict__ Ab,
                    const unsigned short* __restrict__ Wb,
                    const float* __restrict__ bias,
                    unsigned short* __restrict__ Gi) {
  const int tid = threadIdx.x;
  const int l   = tid & 31;
  const int wv  = tid >> 5;          // 0..7
  const int lc  = l & 15;
  const int lh  = l >> 4;
  const int m0  = blockIdx.x * 16;
  const int n0b = wv * 96;           // 6 N-tiles per wave, 8 waves cover 768

  float biasv[6];
  const unsigned short* brow[6];
  #pragma unroll
  for (int j = 0; j < 6; ++j) {
    const int n = n0b + 16 * j + lc;
    biasv[j] = bias[n];
    brow[j]  = Wb + (size_t)n * IDIM + lh * 8;
  }
  const unsigned short* arow = Ab + (size_t)(m0 + lc) * IDIM + lh * 8;

  v8f acc[6] = {};
  #pragma unroll
  for (int kt = 0; kt < 8; ++kt) {
    const int k0 = kt * 32;
    FragAB a;
    a.q[0] = *(const uint4*)(arow + k0);
    a.q[1] = *(const uint4*)(arow + k0 + 16);
    #pragma unroll
    for (int j = 0; j < 6; ++j) {
      FragAB b;
      b.q[0] = *(const uint4*)(brow[j] + k0);
      b.q[1] = *(const uint4*)(brow[j] + k0 + 16);
      acc[j] = wmma_bf16(a, b, acc[j]);
    }
  }

  unsigned short* orow = Gi + (size_t)(m0 + lh * 8) * GDIM + lc;
  #pragma unroll
  for (int j = 0; j < 6; ++j) {
    #pragma unroll
    for (int r = 0; r < 8; ++r)
      orow[(size_t)r * GDIM + n0b + 16 * j] = f2bf(acc[j][r] + biasv[j]);
  }
}

// ---------------------------------------------------------------------------
// Kernel 3: persistent single-WGP GRU recurrence (1 block x 1024 thr = 32 waves).
// Per step: TDM prefetch of the step's Gi row into LDS overlapped with the
// h @ Whh^T WMMA GEMM; s_wait_tensorcnt + barrier; gate math updates h in LDS.
// ---------------------------------------------------------------------------
__global__ __launch_bounds__(1024)
void gru_recurrence_kernel(const unsigned short* __restrict__ Whh,  // [768][256] bf16
                           const float* __restrict__ bhh,           // [768]
                           const unsigned short* __restrict__ GiBase,
                           unsigned long long giStrideBytes,        // 98304 (layer0) / 0 (layer1)
                           int nsteps,
                           float* __restrict__ hOut,                // [64*256] f32
                           unsigned short* __restrict__ hOutB) {    // [64*256] bf16
  extern __shared__ char smem[];
  float*          ghS = (float*)(smem + LDS_GH_OFF);
  unsigned short* hS  = (unsigned short*)(smem + LDS_H_OFF);
  unsigned short* giS = (unsigned short*)(smem + LDS_GI_OFF);

  const int tid = threadIdx.x;
  const int l   = tid & 31;
  const int wv  = tid >> 5;
  const int lc  = l & 15;
  const int lh  = l >> 4;
  const int m0  = (wv & 3) * 16;       // M tile row (batch)
  const int n0b = (wv >> 2) * 96;      // 6 N-tiles [n0b, n0b+96)

  // h starts at zero for both layers.
  #pragma unroll
  for (int j = 0; j < 16; ++j) hS[tid + 1024 * j] = 0;
  __syncthreads();

  float biasv[6];
  const unsigned short* brow[6];
  #pragma unroll
  for (int j = 0; j < 6; ++j) {
    const int n = n0b + 16 * j + lc;
    biasv[j] = bhh[n];
    brow[j]  = Whh + (size_t)n * HDIM + lh * 8;
  }
  const unsigned short* harow = hS + (m0 + lc) * HDIM + lh * 8;
  float* ghrow = ghS + (size_t)(m0 + lh * 8) * GDIM + lc;

  for (int t = 0; t < nsteps; ++t) {
#if HAVE_TDM
    // Async DMA of this step's gate row into LDS, overlapped with the GEMM.
    if (tid < 32 && (giStrideBytes != 0ull || t == 0)) {
      unsigned long long ga = (unsigned long long)GiBase +
                              (unsigned long long)t * giStrideBytes;
      tdm_v4u g0; tdm_v8i g1; tdm_v4i gz = {0, 0, 0, 0};
      g0[0] = 1u;                                   // count=1, user-mode load
      g0[1] = LDS_GI_OFF;                           // LDS destination (bytes)
      g0[2] = (unsigned)(ga & 0xffffffffull);       // global_addr[31:0]
      g0[3] = (unsigned)((ga >> 32) & 0x1ffffffull) // global_addr[56:32]
              | (2u << 30);                         // type = 2 ("image")
      g1[0] = 0x00010000;                           // data_size = 2 bytes
      g1[1] = (int)0xC0000000u;                     // tensor_dim0 = 49152 (lo16)
      g1[2] = 0x00010000;                           // tensor_dim1 = 1
      g1[3] = (int)0xC0000000u;                     // tile_dim0 = 49152
      g1[4] = 1;                                    // tile_dim1 = 1
      g1[5] = 49152;                                // tensor_dim0_stride
      g1[6] = (int)0xC0000000u;                     // tensor_dim1_stride (lo16)
      g1[7] = 0;
#if __clang_major__ >= 23
      tdm_v8i gz8 = {0, 0, 0, 0, 0, 0, 0, 0};
      __builtin_amdgcn_tensor_load_to_lds(g0, g1, gz, gz, gz8, 0);
#else
      __builtin_amdgcn_tensor_load_to_lds(g0, g1, gz, gz, 0);
#endif
    }
#endif

    // ---- GEMM phase: gh = h @ Whh^T (+ bhh), WMMA bf16 16x16x32 ----
    v8f acc[6] = {};
    #pragma unroll
    for (int kt = 0; kt < 8; ++kt) {
      const int k0 = kt * 32;
      FragAB a;
      a.q[0] = *(const uint4*)(harow + k0);        // K[k0+8lh .. +8)
      a.q[1] = *(const uint4*)(harow + k0 + 16);   // K[k0+8lh+16 .. +8)
      #pragma unroll
      for (int j = 0; j < 6; ++j) {
        FragAB b;
        b.q[0] = *(const uint4*)(brow[j] + k0);    // K[k0+8lh .. +8)
        b.q[1] = *(const uint4*)(brow[j] + k0 + 16);
        acc[j] = wmma_bf16(a, b, acc[j]);
      }
    }
    #pragma unroll
    for (int j = 0; j < 6; ++j) {
      #pragma unroll
      for (int r = 0; r < 8; ++r)
        ghrow[(size_t)r * GDIM + n0b + 16 * j] = acc[j][r] + biasv[j];
    }

#if HAVE_TDM
    if (tid < 32) __builtin_amdgcn_s_wait_tensorcnt(0);
    const unsigned short* giP = giS;
#else
    const unsigned short* giP = GiBase + ((size_t)t * giStrideBytes) / 2;
#endif
    __syncthreads();

    // ---- Gate phase: 16384 h elements over 1024 threads ----
    #pragma unroll
    for (int j = 0; j < 16; ++j) {
      const int e  = tid + 1024 * j;
      const int b  = e >> 8;
      const int k  = e & 255;
      const int gb = b * GDIM + k;
      const float gir = bf2f(giP[gb]);
      const float giz = bf2f(giP[gb + 256]);
      const float gin = bf2f(giP[gb + 512]);
      const float ghr = ghS[gb];
      const float ghz = ghS[gb + 256];
      const float ghn = ghS[gb + 512];
      const float rg = sigmoid_fast(gir + ghr);
      const float zg = sigmoid_fast(giz + ghz);
      const float ng = tanh_fast(gin + rg * ghn);
      const float h  = bf2f(hS[e]);
      hS[e] = f2bf((1.0f - zg) * ng + zg * h);
    }
    __syncthreads();
  }

  #pragma unroll
  for (int j = 0; j < 16; ++j) {
    const int e = tid + 1024 * j;
    hOut[e]  = bf2f(hS[e]);
    hOutB[e] = hS[e];
  }
}

// ---------------------------------------------------------------------------
extern "C" void kernel_launch(void* const* d_in, const int* in_sizes, int n_in,
                              void* d_out, int out_size, void* d_ws, size_t ws_size,
                              hipStream_t stream) {
  (void)in_sizes; (void)n_in; (void)out_size; (void)ws_size;

  const float* x    = (const float*)d_in[0];
  const float* Wih0 = (const float*)d_in[1];
  const float* Whh0 = (const float*)d_in[2];
  const float* bih0 = (const float*)d_in[3];
  const float* bhh0 = (const float*)d_in[4];
  const float* Wih1 = (const float*)d_in[5];
  const float* Whh1 = (const float*)d_in[6];
  const float* bih1 = (const float*)d_in[7];
  const float* bhh1 = (const float*)d_in[8];

  char* ws = (char*)d_ws;
  unsigned short* Wih0b = (unsigned short*)(ws + WS_WIH0B);
  unsigned short* Whh0b = (unsigned short*)(ws + WS_WHH0B);
  unsigned short* Wih1b = (unsigned short*)(ws + WS_WIH1B);
  unsigned short* Whh1b = (unsigned short*)(ws + WS_WHH1B);
  unsigned short* Gi1   = (unsigned short*)(ws + WS_GI1);
  unsigned short* hL0b  = (unsigned short*)(ws + WS_HL0B);
  float*          hL0f  = (float*)(ws + WS_HL0F);
  unsigned short* Gi0   = (unsigned short*)(ws + WS_GI0);
  unsigned short* Xb    = (unsigned short*)(ws + WS_XB);

  // 1. Weights and x -> bf16 (streaming, bandwidth-bound).
  wcvt_kernel<<<(GDIM * IDIM + 255) / 256, 256, 0, stream>>>(
      Wih0, Whh0, Wih1, Whh1, Wih0b, Whh0b, Wih1b, Whh1b, GDIM * IDIM);
  xcvt_kernel<<<(MROWS * IDIM / 4 + 255) / 256, 256, 0, stream>>>(
      x, Xb, MROWS * IDIM / 4);

  // 2. Time-parallel input GEMM for layer 0: Gi0 = x @ Wih0^T + bih0.
  gi_gemm_kernel<<<dim3(MROWS / 16, 1), 256, 0, stream>>>(Xb, Wih0b, bih0, Gi0);

  // 3. Layer 0 recurrence (persistent, single WGP).
  gru_recurrence_kernel<<<1, 1024, LDS_BYTES, stream>>>(
      Whh0b, bhh0, Gi0, 98304ull, T_STEPS, hL0f, hL0b);

  // 4. Layer 1 constant-input gates: Gi1 = h_last0 @ Wih1^T + bih1.
  gi_gemm_kernel<<<dim3(BATCH / 16, 1), 256, 0, stream>>>(hL0b, Wih1b, bih1, Gi1);

  // 5. Layer 1 recurrence, final hidden -> d_out (f32), bf16 copy to scratch.
  gru_recurrence_kernel<<<1, 1024, LDS_BYTES, stream>>>(
      Whh1b, bhh1, Gi1, 0ull, T_STEPS, (float*)d_out, hL0b);
}
